// fratFace_52261162057741
// MI455X (gfx1250) — matrix operand order repaired and checked by reading, not compiled, not run on hardware
//
#include <hip/hip_runtime.h>
#include <math.h>

#define B_SZ 512
#define E_SZ 512
#define C_SZ 70722
#define M_MARGIN 0.4f
#define H_CONST 0.333f
#define S_SCALE 64.0f
#define T_ALPHA 1.0f
#define EPS_F 1e-3f
#define PI_F 3.14159265358979323846f

typedef __attribute__((ext_vector_type(16))) __bf16 v16bf;
typedef __attribute__((ext_vector_type(8)))  float  v8f;

#if defined(__AMDGCN__) && __has_builtin(__builtin_amdgcn_tensor_load_to_lds)
#define USE_TDM 1
#else
#define USE_TDM 0
#endif

// Pack two fp32 -> two bf16 (round-half-up) in one v_perm_b32.
// v_perm_b32 selects bytes from {S0(bytes 4-7), S1(bytes 0-3)}:
// sel 0x07060302 -> dst = {S0.b3, S0.b2, S1.b3, S1.b2} = hi_bf16<<16 | lo_bf16
__device__ __forceinline__ unsigned pack_bf2(float lo, float hi) {
    union { float f; unsigned u; } a, b;
    a.f = lo; b.f = hi;
    unsigned au = a.u + 0x8000u;
    unsigned bu = b.u + 0x8000u;
    return __builtin_amdgcn_perm(bu, au, 0x07060302u);
}

#if USE_TDM
typedef unsigned int u32x4 __attribute__((ext_vector_type(4)));
typedef int          i32x8 __attribute__((ext_vector_type(8)));
typedef int          i32x4 __attribute__((ext_vector_type(4)));

// TDM: DMA a 32-row x 256-col fp32 tile (row stride 2*C elements) from global
// into LDS. pad: 1 dword after every 256 dwords -> LDS pitch of 257 dwords,
// which makes the later column-major (transposing) LDS reads bank-conflict-free.
__device__ __forceinline__ void tdm_load_btile(const float* gsrc, unsigned lds_off,
                                               unsigned cols_rem) {
    unsigned long long ga = (unsigned long long)(size_t)gsrc;
    u32x4 g0;
    g0[0] = 1u;                                        // count=1, user descriptor
    g0[1] = lds_off;                                   // LDS byte address
    g0[2] = (unsigned)(ga & 0xFFFFFFFFu);              // global_addr[31:0]
    g0[3] = (unsigned)((ga >> 32) & 0x01FFFFFFu)       // global_addr[56:32]
          | (2u << 30);                                // type = 2 ("image")
    i32x8 g1;
    const unsigned td0 = cols_rem;                     // tensor_dim0 (elements)
    const unsigned td1 = (unsigned)E_SZ;               // tensor_dim1
    g1[0] = (int)((2u << 16)                           // data_size = 4 bytes
                | (1u << 20)                           // pad_enable
                | (7u << 22)                           // pad_interval = 256 dwords
                | (0u << 25));                         // pad_amount = 1 dword
    g1[1] = (int)((td0 & 0xFFFFu) << 16);              // td0[15:0] at bits 63:48
    g1[2] = (int)(((td0 >> 16) & 0xFFFFu)              // td0[31:16]
                | ((td1 & 0xFFFFu) << 16));            // td1[15:0]
    g1[3] = (int)(((td1 >> 16) & 0xFFFFu)              // td1[31:16]
                | (256u << 16));                       // tile_dim0 = 256
    g1[4] = 32;                                        // tile_dim1 = 32, tile_dim2 = 0
    g1[5] = (int)(2u * C_SZ);                          // tensor_dim0_stride lo32
    g1[6] = 0;                                         // stride0 hi16 | stride1 lo16
    g1[7] = 0;
    i32x4 gz4 = {0, 0, 0, 0};                          // groups 2/3: <=2D tensor
    i32x8 gz8 = {0, 0, 0, 0, 0, 0, 0, 0};              // trailing group (6-arg form)
    // This toolchain exposes the 6-arg builtin:
    // (g0:u32x4, g1:i32x8, g2:i32x4, g3:i32x4, g4:i32x8, cpol:i32)
    __builtin_amdgcn_tensor_load_to_lds(g0, g1, gz4, gz4, gz8, 0);
}
#endif

// ---------------------------------------------------------------------------
// Kernel A: inverse column norms of the t=0 plane of `kernel` (E x 2 x C).
// ---------------------------------------------------------------------------
__global__ __launch_bounds__(256) void col_norms_t0(const float* __restrict__ kern,
                                                    float* __restrict__ invn0) {
    int c = blockIdx.x * blockDim.x + threadIdx.x;
    if (c >= C_SZ) return;
    const float* p = kern + c;
    float ss = 0.f;
    #pragma unroll 8
    for (int e = 0; e < E_SZ; ++e) {
        float v = p[(size_t)e * (2 * C_SZ)];
        ss += v * v;
    }
    invn0[c] = rsqrtf(ss);
}

// ---------------------------------------------------------------------------
// Kernel B: batch statistics, per-row margin scalars, t=1 target dot products.
// Single block of 512 threads (one thread per batch row).
// ---------------------------------------------------------------------------
__global__ __launch_bounds__(512) void stats_kernel(const float* __restrict__ emb,
                                                    const float* __restrict__ norms,
                                                    const int* __restrict__ label,
                                                    const float* __restrict__ kern,
                                                    float* __restrict__ out,
                                                    float* __restrict__ cos2t,
                                                    float* __restrict__ g_ang,
                                                    float* __restrict__ g_add) {
    __shared__ float s_sn[B_SZ];
    __shared__ int   s_lab[B_SZ];
    __shared__ float red[B_SZ];

    const int b = threadIdx.x;
    float sn = fminf(fmaxf(norms[b], 0.001f), 100.0f);
    s_sn[b]  = sn;
    s_lab[b] = label[b];
    __syncthreads();

    // sum reduction
    red[b] = sn; __syncthreads();
    for (int s = B_SZ / 2; s > 0; s >>= 1) {
        if (b < s) red[b] += red[b + s];
        __syncthreads();
    }
    float total = red[0];
    __syncthreads();
    // sum of squares reduction
    red[b] = sn * sn; __syncthreads();
    for (int s = B_SZ / 2; s > 0; s >>= 1) {
        if (b < s) red[b] += red[b + s];
        __syncthreads();
    }
    float sumsq = red[0];

    float mean = total / (float)B_SZ;
    float var  = (sumsq - total * mean) / (float)(B_SZ - 1);   // ddof = 1
    float stdv = sqrtf(fmaxf(var, 0.f));

    float bmean = mean * T_ALPHA + (1.0f - T_ALPHA) * 20.0f;
    float bstd  = stdv * T_ALPHA + (1.0f - T_ALPHA) * 10.0f;

    // class-wise mean via O(B^2) scan over shared label/sn (trivial at B=512)
    const int myl = s_lab[b];
    float cnt = 0.f, csum = 0.f;
    for (int j = 0; j < B_SZ; ++j) {
        if (s_lab[j] == myl) { cnt += 1.f; csum += s_sn[j]; }
    }
    float cwmean = (cnt > 0.f)
        ? (csum / fmaxf(cnt, 1.f)) * T_ALPHA + (1.0f - T_ALPHA) * 20.0f
        : 20.0f;

    float ms = fminf(fmaxf((sn - bmean) / (bstd + EPS_F) * H_CONST, -1.f), 1.f);
    float cw = fminf(fmaxf((sn - cwmean) / (bstd + EPS_F), -1.f), 1.f) * H_CONST;

    out[(size_t)2 * B_SZ * C_SZ + b] = cw;     // cw_margin_scaler output
    g_ang[b] = -M_MARGIN * ms;                 // origin_m_arc at target
    g_add[b] = M_MARGIN + M_MARGIN * ms;       // origin_m_cos at target

    // cos2_tgt[b] = emb[b,:] . normalize(kernel[:,1,label[b]])
    const float* kcol = kern + C_SZ + myl;         // t=1 plane
    const float* erow = emb + (size_t)b * E_SZ;
    float ss = 0.f, dot = 0.f;
    #pragma unroll 4
    for (int e = 0; e < E_SZ; ++e) {
        float kv = kcol[(size_t)e * (2 * C_SZ)];
        ss  += kv * kv;
        dot += erow[e] * kv;
    }
    float c2 = dot * rsqrtf(ss);
    cos2t[b] = fminf(fmaxf(c2, -1.0f + EPS_F), 1.0f - EPS_F);
}

// ---------------------------------------------------------------------------
// Kernel C: bf16 WMMA GEMM over the t=0 plane + clip/scale epilogue.
// Block tile 32(M) x 256(N), K-step 32; 8 waves, each owning a 16x64 strip.
// B tile moved by the Tensor Data Mover (global -> LDS), then transposed +
// converted to bf16 through LDS.
// ---------------------------------------------------------------------------
#define LDS_PITCH 40   // 32 + 8 halves: 80B rows -> 16B aligned, 20-bank stride
#define RAW_PITCH 257  // fp32 dwords per TDM row (pad gives conflict-free columns)

__global__ __launch_bounds__(256) void gemm_epilogue(const float* __restrict__ emb,
                                                     const float* __restrict__ kern,
                                                     const float* __restrict__ invn0,
                                                     float* __restrict__ out) {
    __shared__ unsigned short lds_a[32 * LDS_PITCH];
    __shared__ unsigned short lds_b[256 * LDS_PITCH];
#if USE_TDM
    __shared__ float lds_rawf[32 * RAW_PITCH];
#endif

    const int tid   = threadIdx.x;
    const int lane  = tid & 31;
    const int wid   = tid >> 5;     // 0..7
    const int waveM = wid >> 2;     // 0..1
    const int waveN = wid & 3;      // 0..3
    const int mBase = blockIdx.x * 32;
    const int cBase = blockIdx.y * 256;

    v8f acc[4] = {};

    // A staging indices: 32 rows x 32 k, one float4 per thread
    const int a_row = tid >> 3;
    const int a_k   = (tid & 7) * 4;
#if USE_TDM
    const unsigned lds_raw_off = (unsigned)(size_t)(void*)&lds_rawf[0];
#else
    // Fallback staging: one column per thread; clamp (not predicate) the column
    // so loads stay unconditional -- garbage only lands in columns >= C, which
    // are never stored (WMMA does not mix data across N).
    const int cg  = cBase + tid;
    const int cgc = (cg < C_SZ) ? cg : (C_SZ - 1);
#endif

    for (int kb = 0; kb < E_SZ; kb += 32) {
#if USE_TDM
        // ---- TDM: DMA the 32x256 fp32 B tile into LDS (pitch 257 dwords).
        // Only wave 0 issues/waits; previous-iteration readers of lds_rawf
        // finished before this wave passed the last barrier.
        if (tid < 32) {
            tdm_load_btile(kern + (size_t)kb * (2 * C_SZ) + cBase,
                           lds_raw_off, (unsigned)(2 * C_SZ - cBase));
            __builtin_amdgcn_s_wait_tensorcnt(0);
        }
        __syncthreads();   // publish tile; also fences prior lds_b consumers
#endif
        // ---- stage A tile (fp32 -> packed bf16)
        {
            const float4 av = *(const float4*)(emb + (size_t)(mBase + a_row) * E_SZ + kb + a_k);
            uint2 pa;
            pa.x = pack_bf2(av.x, av.y);
            pa.y = pack_bf2(av.z, av.w);
            *(uint2*)&lds_a[a_row * LDS_PITCH + a_k] = pa;
        }
        // ---- stage B tile: transpose to [c][k] + convert to packed bf16
        {
            unsigned pk[16];
#if USE_TDM
            #pragma unroll
            for (int k = 0; k < 32; k += 2) {
                float v0 = lds_rawf[(k)     * RAW_PITCH + tid];
                float v1 = lds_rawf[(k + 1) * RAW_PITCH + tid];
                pk[k >> 1] = pack_bf2(v0, v1);
            }
#else
            const float* src = kern + (size_t)kb * (2 * C_SZ) + cgc;
            #pragma unroll
            for (int k = 0; k < 32; k += 2) {
                float v0 = src[(size_t)k       * (2 * C_SZ)];
                float v1 = src[(size_t)(k + 1) * (2 * C_SZ)];
                pk[k >> 1] = pack_bf2(v0, v1);
            }
            // prefetch next K-step's B rows (global_prefetch_b8)
            if (kb + 32 < E_SZ) {
                const float* nsrc = kern + (size_t)(kb + 32) * (2 * C_SZ) + cgc;
                #pragma unroll
                for (int k = 0; k < 32; k += 8)
                    __builtin_prefetch(nsrc + (size_t)k * (2 * C_SZ), 0, 3);
            }
#endif
            uint4* dst = (uint4*)&lds_b[tid * LDS_PITCH];
            const uint4* s = (const uint4*)pk;
            dst[0] = s[0]; dst[1] = s[1]; dst[2] = s[2]; dst[3] = s[3];
        }
        __syncthreads();

        // ---- fragments + WMMA
        {
            const int hlf  = lane >> 4;
            const int lrow = lane & 15;
            union { v16bf v; uint4 q[2]; } afrag;
            // A k order used here (same as B): half 0 -> K{0..15}, half 1 -> K{16..31}
            const unsigned short* ap = &lds_a[(waveM * 16 + lrow) * LDS_PITCH];
            afrag.q[0] = *(const uint4*)(ap + hlf * 16);
            afrag.q[1] = *(const uint4*)(ap + hlf * 16 + 8);
            #pragma unroll
            for (int s4 = 0; s4 < 4; ++s4) {
                union { v16bf v; uint4 q[2]; } bfrag;
                // B k order used here (same as A): lanes 0-15 hold K=0..15, lanes 16-31 K=16..31
                const unsigned short* bp =
                    &lds_b[(waveN * 64 + s4 * 16 + lrow) * LDS_PITCH + hlf * 16];
                bfrag.q[0] = ((const uint4*)bp)[0];
                bfrag.q[1] = ((const uint4*)bp)[1];
                acc[s4] = __builtin_amdgcn_wmma_f32_16x16x32_bf16(
                    false, afrag.v, false, bfrag.v, (short)0, acc[s4], false, false);
            }
        }
        __syncthreads();
    }

    // ---- epilogue: scale by inv column norm, clip, *S, write both planes
    const int hlf  = lane >> 4;
    const int lcol = lane & 15;
    #pragma unroll
    for (int s4 = 0; s4 < 4; ++s4) {
        int col = cBase + waveN * 64 + s4 * 16 + lcol;
        if (col < C_SZ) {
            float invn = invn0[col];
            #pragma unroll
            for (int j = 0; j < 8; ++j) {
                int row = mBase + waveM * 16 + hlf * 8 + j;   // C/D VGPR layout
                float w = acc[s4][j] * invn;
                w = fminf(fmaxf(w, -1.0f + EPS_F), 1.0f - EPS_F);
                float o = w * S_SCALE;                         // off-target closed form
                size_t idx = (size_t)row * C_SZ + col;
                __builtin_nontemporal_store(o, out + idx);
                __builtin_nontemporal_store(o, out + (size_t)B_SZ * C_SZ + idx);
            }
        }
    }
}

// ---------------------------------------------------------------------------
// Kernel D: apply full arccos/cos margin formula at the B target positions.
// ---------------------------------------------------------------------------
__global__ void fixup_targets(const int* __restrict__ label,
                              const float* __restrict__ cos2t,
                              const float* __restrict__ g_ang,
                              const float* __restrict__ g_add,
                              float* __restrict__ out) {
    int b = blockIdx.x * blockDim.x + threadIdx.x;
    if (b >= B_SZ) return;
    int c = label[b];
    float ga = g_ang[b], gc = g_add[b];
    size_t idx = (size_t)b * C_SZ + c;

    float w1 = out[idx] * (1.0f / S_SCALE);   // recover clipped cosine (S = 2^6, exact)
    float th = fminf(fmaxf(acosf(w1) + ga, EPS_F), PI_F - EPS_F);
    out[idx] = (cosf(th) - gc) * S_SCALE;

    float w2  = cos2t[b];
    float th2 = fminf(fmaxf(acosf(w2) + ga, EPS_F), PI_F - EPS_F);
    out[(size_t)B_SZ * C_SZ + idx] = (cosf(th2) - gc) * S_SCALE;
}

// ---------------------------------------------------------------------------
extern "C" void kernel_launch(void* const* d_in, const int* in_sizes, int n_in,
                              void* d_out, int out_size, void* d_ws, size_t ws_size,
                              hipStream_t stream) {
    (void)in_sizes; (void)n_in; (void)out_size; (void)ws_size;
    const float* emb   = (const float*)d_in[0];
    const float* norms = (const float*)d_in[1];
    const int*   label = (const int*)d_in[2];
    const float* kern  = (const float*)d_in[3];
    float* out = (float*)d_out;

    float* ws    = (float*)d_ws;
    float* invn0 = ws;                 // C floats
    float* cos2t = invn0 + C_SZ;       // B floats
    float* g_ang = cos2t + B_SZ;       // B floats
    float* g_add = g_ang + B_SZ;       // B floats

    dim3 gA((C_SZ + 255) / 256);
    col_norms_t0<<<gA, 256, 0, stream>>>(kern, invn0);

    stats_kernel<<<1, 512, 0, stream>>>(emb, norms, label, kern, out, cos2t, g_ang, g_add);

    // M on the fast grid axis so m-blocks sharing a B tile are L2-adjacent.
    dim3 gC(B_SZ / 32, (C_SZ + 255) / 256);
    gemm_epilogue<<<gC, 256, 0, stream>>>(emb, kern, invn0, out);

    fixup_targets<<<(B_SZ + 255) / 256, 256, 0, stream>>>(label, cos2t, g_ang, g_add, out);
}
